// BiLSTM_CRF_25958782337473
// MI455X (gfx1250) — matrix operand order repaired and checked by reading, not compiled, not run on hardware
//
#include <hip/hip_runtime.h>
#include <hip/hip_bf16.h>

// ---------------------------------------------------------------------------
// BiLSTM-CRF for MI455X (gfx1250, wave32).
//  - x@Wih^T hoisted into a full-chip WMMA bf16 GEMM (Xg), stored
//    column-major [t][gatecol][batch] -> recurrence C-init is one b128/lane.
//  - Recurrence: 8 batch-slices x 2 directions = 16 workgroups (8 waves).
//    Whh async-staged to LDS once, then held in REGISTERS per wave (16 B
//    fragments); h-slice double-buffered in LDS (1 barrier/step); cell state
//    and previous-h in VGPRs; next-step Xg loads software-pipelined.
//  - Hs stored column-major [t][j][batch] -> coalesced feats kernel.
// ---------------------------------------------------------------------------

typedef __attribute__((ext_vector_type(16))) __bf16 v16bf;
typedef __attribute__((ext_vector_type(8)))  float  v8f;

#define T_LEN 512
#define BATCH 128
#define DIM   256
#define HHID  128
#define G4H   512

// ---- WMMA fragment loaders (CDNA5 ISA 7.12.2 layouts, wave32) -------------

// A: 16x32 bf16 tile (M x K), row-major source with leading dim `ld`.
__device__ __forceinline__ v16bf load_a_bf16(const __bf16* tile, int ld, int lane) {
  const int row = lane & 15;
  const int grp = lane >> 4;
  v16bf a;
#pragma unroll
  for (int e = 0; e < 16; ++e) {
    const int vg = e >> 1, sub = e & 1;
    const int k = (vg < 4) ? (vg * 2 + sub + grp * 8)
                           : (16 + (vg - 4) * 2 + sub + grp * 8);
    a[e] = tile[row * ld + k];
  }
  return a;
}

// B: 32x16 bf16 tile where B[k][n] = W[nbase+n][kbase+k], W row-major [N][ldk].
__device__ __forceinline__ v16bf load_b_wT(const __bf16* W, int ldk,
                                           int nbase, int kbase, int lane) {
  const int n = lane & 15;
  const int grp = lane >> 4;
  const __bf16* p = W + (size_t)(nbase + n) * ldk + kbase + grp * 8;
  v16bf b;
#pragma unroll
  for (int e = 0; e < 16; ++e) b[e] = p[e + (e >> 3) * 8];
  return b;
}

// ---- fp32 -> bf16 weight conversion ---------------------------------------
__global__ void cvt_bf16_kernel(const float* __restrict__ s,
                                __bf16* __restrict__ d, int n) {
  const int i = blockIdx.x * blockDim.x + threadIdx.x;
  if (i < n) d[i] = (__bf16)s[i];
}

// ---- Xg = gather(emb)@Wih^T + bias (bf16 WMMA, f32 accum) -----------------
// Output column-major: Xg[t][col (512)][batch (128)] bf16.
__global__ __launch_bounds__(256) void xgemm_kernel(
    const int* __restrict__ sentence, const float* __restrict__ emb,
    const __bf16* __restrict__ Wih, const float* __restrict__ bias,
    __bf16* __restrict__ Xg) {
  __shared__ __bf16 Ash[16][DIM];   // 8 KB: 16 gathered embedding rows

  const int tid = threadIdx.x;
  const int rbase = blockIdx.x * 16;          // row = t*128 + b
  {
    const int row = tid >> 4;
    const int cb  = (tid & 15) * 16;
    const int rglob = rbase + row;
    const int t = rglob >> 7;
    const int b = rglob & 127;
    const int tok = sentence[b * T_LEN + t];
    const float* src = emb + (size_t)tok * DIM + cb;
#pragma unroll
    for (int i = 0; i < 16; ++i) Ash[row][cb + i] = (__bf16)src[i];
  }
  __syncthreads();

  const int wave = tid >> 5, lane = tid & 31;
  const int nbase = (blockIdx.y * 8 + wave) * 16;
  const int ncol  = nbase + (lane & 15);
  const int grp   = lane >> 4;

  v8f acc;
  const float bn = bias[ncol];
#pragma unroll
  for (int r = 0; r < 8; ++r) acc[r] = bn;

#pragma unroll
  for (int ks = 0; ks < 8; ++ks) {
    v16bf a = load_a_bf16(&Ash[0][ks * 32], DIM, lane);
    v16bf b = load_b_wT(Wih, DIM, nbase, ks * 32, lane);
    acc = __builtin_amdgcn_wmma_f32_16x16x32_bf16(false, a, false, b,
                                                  (short)0, acc, false, false);
  }
  const int t  = rbase >> 7;
  const int bb = (rbase & 127) + 8 * grp;
  union { uint4 u; __bf16 h[8]; } pk;
#pragma unroll
  for (int r = 0; r < 8; ++r) pk.h[r] = (__bf16)acc[r];
  *(uint4*)&Xg[((size_t)t * G4H + ncol) * BATCH + bb] = pk.u;
}

// ---- LSTM recurrence: 8 batch-slices x 2 directions -----------------------
__global__ __launch_bounds__(256) void lstm_kernel(
    const __bf16* __restrict__ XgF, const __bf16* __restrict__ XgB,
    const __bf16* __restrict__ WhhF, const __bf16* __restrict__ WhhB,
    const int* __restrict__ lengths,
    __bf16* __restrict__ HsF, __bf16* __restrict__ HsB) {
  __shared__ __bf16 Wsh[G4H * HHID];      // 128 KB, async-staged once
  __shared__ __bf16 Hsh[2][16 * HHID];    //   8 KB double-buffered h slice

  const int tid = threadIdx.x;
  const int bslice = blockIdx.x;          // 0..7
  const int dir    = blockIdx.y;          // 0 fwd, 1 bwd
  const int b0 = bslice * 16;
  const __bf16* Xg  = dir ? XgB : XgF;
  const __bf16* Whh = dir ? WhhB : WhhF;
  __bf16*       Hs  = dir ? HsB : HsF;

  // one-time async fill of Whh into LDS (tracked on ASYNCcnt)
  {
    const char* src = (const char*)Whh;
    for (int i = tid; i < (G4H * HHID * 2) / 16; i += 256) {
      const uint32_t dst = (uint32_t)(uintptr_t)(&Wsh[0]) + (uint32_t)(i * 16);
      const void* g = src + (size_t)i * 16;
      asm volatile("global_load_async_to_lds_b128 %0, %1, off"
                   :: "v"(dst), "v"(g) : "memory");
    }
  }
  for (int i = tid; i < 16 * HHID; i += 256) Hsh[0][i] = (__bf16)0.0f;
  asm volatile("s_wait_asynccnt 0" ::: "memory");
  __syncthreads();

  const int wave = tid >> 5, lane = tid & 31;
  const int n = lane & 15, grp = lane >> 4;
  const int jt = wave;
  const int j  = jt * 16 + n;

  // hoist this wave's 16 Whh B-fragments into registers (constant over time)
  v16bf bfr[4][4];
#pragma unroll
  for (int ch = 0; ch < 4; ++ch)
#pragma unroll
    for (int ks = 0; ks < 4; ++ks)
      bfr[ch][ks] = load_b_wT(Wsh, HHID, ch * HHID + jt * 16, ks * 32, lane);

  v8f c;
  float hq[8];                 // previous h for this wave's elements
  int lenr[8];
#pragma unroll
  for (int r = 0; r < 8; ++r) {
    c[r] = 0.0f;
    hq[r] = 0.0f;
    lenr[r] = lengths[b0 + r + 8 * grp];
  }

  // prime the Xg software pipeline for step 0
  const size_t xoff = (size_t)b0 + 8 * grp;
  uint4 xg[4];
  {
    const int t0 = dir ? (T_LEN - 1) : 0;
#pragma unroll
    for (int ch = 0; ch < 4; ++ch)
      xg[ch] = *(const uint4*)&Xg[((size_t)t0 * G4H + ch * HHID + j) * BATCH + xoff];
  }

  for (int s = 0; s < T_LEN; ++s) {
    const int t = dir ? (T_LEN - 1 - s) : s;
    const int cur = s & 1, nxt = cur ^ 1;

    v16bf afr[4];
#pragma unroll
    for (int ks = 0; ks < 4; ++ks)
      afr[ks] = load_a_bf16(&Hsh[cur][ks * 32], HHID, lane);

    // issue next step's Xg loads now (clamped; overlaps WMMAs + activations)
    const int sn = (s + 1 < T_LEN) ? (s + 1) : s;
    const int t2 = dir ? (T_LEN - 1 - sn) : sn;
    uint4 xgn[4];
#pragma unroll
    for (int ch = 0; ch < 4; ++ch)
      xgn[ch] = *(const uint4*)&Xg[((size_t)t2 * G4H + ch * HHID + j) * BATCH + xoff];
    if (s + 2 < T_LEN) {
      const int t3 = dir ? (T_LEN - 3 - s) : (s + 2);
      __builtin_prefetch(&Xg[((size_t)t3 * G4H + j) * BATCH + xoff], 0, 1);
    }

    v8f g[4];
#pragma unroll
    for (int ch = 0; ch < 4; ++ch) {
      union { uint4 u; __bf16 h[8]; } pk;
      pk.u = xg[ch];
      v8f acc;
#pragma unroll
      for (int r = 0; r < 8; ++r) acc[r] = (float)pk.h[r];
#pragma unroll
      for (int ks = 0; ks < 4; ++ks)
        acc = __builtin_amdgcn_wmma_f32_16x16x32_bf16(
            false, afr[ks], false, bfr[ch][ks], (short)0, acc, false, false);
      g[ch] = acc;
    }

    union { uint4 u; __bf16 h[8]; } hsv;
#pragma unroll
    for (int r = 0; r < 8; ++r) {
      const float gi = g[0][r], gf = g[1][r], gg = g[2][r], go = g[3][r];
      const float si = 1.0f / (1.0f + __expf(-gi));
      const float sf = 1.0f / (1.0f + __expf(-gf));
      const float so = 1.0f / (1.0f + __expf(-go));
      const float cn = sf * c[r] + si * tanhf(gg);
      const float hn = so * tanhf(cn);
      const bool msk = t < lenr[r];
      const float h2 = msk ? hn : hq[r];   // h_old lives in registers
      c[r]  = msk ? cn : c[r];
      hq[r] = h2;
      hsv.h[r] = (__bf16)h2;
    }
    // column-major global store: 8 contiguous batch entries, one b128
    *(uint4*)&Hs[((size_t)t * HHID + j) * BATCH + xoff] = hsv.u;

    // publish h_new into the other buffer; single barrier per step
#pragma unroll
    for (int r = 0; r < 8; ++r)
      Hsh[nxt][(r + 8 * grp) * HHID + j] = (__bf16)hq[r];
    __syncthreads();

#pragma unroll
    for (int ch = 0; ch < 4; ++ch) xg[ch] = xgn[ch];
  }
}

// ---- feats = concat(hf, hb) @ Wt^T + bt  ----------------------------------
__global__ __launch_bounds__(128) void feats_kernel(
    const __bf16* __restrict__ HsF, const __bf16* __restrict__ HsB,
    const float* __restrict__ Wt, const float* __restrict__ bt,
    float* __restrict__ feats) {
  __shared__ float Wsh[9 * 256];
  __shared__ float bsh[9];
  const int tid = threadIdx.x;
  for (int i = tid; i < 9 * 256; i += 128) Wsh[i] = Wt[i];
  if (tid < 9) bsh[tid] = bt[tid];
  __syncthreads();

  const int t = blockIdx.x;
  const int b = tid;
  float acc[9];
#pragma unroll
  for (int k = 0; k < 9; ++k) acc[k] = bsh[k];
  for (int d = 0; d < HHID; ++d) {
    const float hf = (float)HsF[((size_t)t * HHID + d) * BATCH + b];
#pragma unroll
    for (int k = 0; k < 9; ++k) acc[k] += hf * Wsh[k * 256 + d];
  }
  for (int d = 0; d < HHID; ++d) {
    const float hb = (float)HsB[((size_t)t * HHID + d) * BATCH + b];
#pragma unroll
    for (int k = 0; k < 9; ++k) acc[k] += hb * Wsh[k * 256 + 128 + d];
  }
  float* o = feats + ((size_t)t * BATCH + b) * 9;
#pragma unroll
  for (int k = 0; k < 9; ++k) o[k] = acc[k];
}

// ---- CRF max-plus scan: one lane per batch element ------------------------
__global__ __launch_bounds__(128) void crf_kernel(
    const float* __restrict__ feats, const int* __restrict__ lengths,
    const float* __restrict__ trans, float* __restrict__ out) {
  __shared__ float tr[81];
  if (threadIdx.x < 81) tr[threadIdx.x] = trans[threadIdx.x];
  __syncthreads();

  const int b = threadIdx.x;
  float fv[9], fin[9];
#pragma unroll
  for (int k = 0; k < 9; ++k) { fv[k] = -10000.0f; fin[k] = -10000.0f; }
  fv[7] = 0.0f;  // START
  const int len = lengths[b];

  for (int t = 0; t < T_LEN; ++t) {
    const float* ft = feats + ((size_t)t * BATCH + b) * 9;
    float nf[9];
#pragma unroll
    for (int k = 0; k < 9; ++k) {
      float m = fv[0] + tr[k * 9 + 0];
#pragma unroll
      for (int jj = 1; jj < 9; ++jj) m = fmaxf(m, fv[jj] + tr[k * 9 + jj]);
      nf[k] = m + ft[k];
    }
#pragma unroll
    for (int k = 0; k < 9; ++k) fv[k] = nf[k];
    if (t == len - 1) {
#pragma unroll
      for (int k = 0; k < 9; ++k) fin[k] = fv[k];
    }
  }
  float best = fin[0] + tr[8 * 9 + 0];   // STOP row
#pragma unroll
  for (int k = 1; k < 9; ++k) best = fmaxf(best, fin[k] + tr[8 * 9 + k]);
  out[b] = best;
}

// ---------------------------------------------------------------------------
extern "C" void kernel_launch(void* const* d_in, const int* in_sizes, int n_in,
                              void* d_out, int out_size, void* d_ws, size_t ws_size,
                              hipStream_t stream) {
  const int*   sentence = (const int*)d_in[0];
  const int*   lengths  = (const int*)d_in[1];
  const float* emb      = (const float*)d_in[2];
  const float* WihF     = (const float*)d_in[3];
  const float* WhhF     = (const float*)d_in[4];
  const float* bF       = (const float*)d_in[5];
  const float* WihB     = (const float*)d_in[6];
  const float* WhhB     = (const float*)d_in[7];
  const float* bB       = (const float*)d_in[8];
  const float* Wt       = (const float*)d_in[9];
  const float* btv      = (const float*)d_in[10];
  const float* trans    = (const float*)d_in[11];
  float* out = (float*)d_out;

  char* w = (char*)d_ws;
  __bf16* WihFb = (__bf16*)(w + 0);                           // 512*256*2
  __bf16* WihBb = (__bf16*)(w + 262144);
  __bf16* WhhFb = (__bf16*)(w + 524288);                      // 512*128*2
  __bf16* WhhBb = (__bf16*)(w + 655360);
  __bf16* XgF   = (__bf16*)(w + 786432);                      // 65536*512*2
  __bf16* XgB   = (__bf16*)(w + 786432 + 67108864ull);
  __bf16* HsF   = (__bf16*)(w + 786432 + 2ull * 67108864);    // 65536*128*2
  __bf16* HsB   = (__bf16*)(w + 786432 + 2ull * 67108864 + 16777216ull);
  float*  feats = (float*) (w + 786432 + 2ull * 67108864 + 2ull * 16777216);

  cvt_bf16_kernel<<<(131072 + 255) / 256, 256, 0, stream>>>(WihF, WihFb, 131072);
  cvt_bf16_kernel<<<(131072 + 255) / 256, 256, 0, stream>>>(WihB, WihBb, 131072);
  cvt_bf16_kernel<<<(65536  + 255) / 256, 256, 0, stream>>>(WhhF, WhhFb, 65536);
  cvt_bf16_kernel<<<(65536  + 255) / 256, 256, 0, stream>>>(WhhB, WhhBb, 65536);

  xgemm_kernel<<<dim3(4096, 4), 256, 0, stream>>>(sentence, emb, WihFb, bF, XgF);
  xgemm_kernel<<<dim3(4096, 4), 256, 0, stream>>>(sentence, emb, WihBb, bB, XgB);

  lstm_kernel<<<dim3(8, 2), 256, 0, stream>>>(XgF, XgB, WhhFb, WhhBb,
                                              lengths, HsF, HsB);

  feats_kernel<<<T_LEN, 128, 0, stream>>>(HsF, HsB, Wt, btv, feats);

  crf_kernel<<<1, 128, 0, stream>>>(feats, lengths, trans, out);
}
